// DecoderStack_64647847739884
// MI455X (gfx1250) — matrix operand order repaired and checked by reading, not compiled, not run on hardware
//
#include <hip/hip_runtime.h>

// ---------------------------------------------------------------------------
// Types for CDNA5 WMMA (wave32): A/B = 16 halfs per lane, C/D = 8 floats/lane
// ---------------------------------------------------------------------------
typedef _Float16 v16h __attribute__((ext_vector_type(16)));
typedef _Float16 h8   __attribute__((ext_vector_type(8)));
typedef float    v8f  __attribute__((ext_vector_type(8)));

#define BM 128
#define BN 64
#define BK 32
#define LPAD 8  // LDS row padding (halfs); keeps 16B alignment, spreads banks

// ---------------------------------------------------------------------------
// Generic fp32-in / fp32-out GEMM, f16 WMMA compute, f32 accumulate.
//   C[z] = alpha * A[z] (MxK) * B[z] (KxN or N x K if transB) + bias, relu opt
// Batch index z = blockIdx.z decomposed as z1 = z % Z1, z2 = z / Z1 with
// independent strides for A/B/C, so one kernel serves per-(head,batch) GEMMs.
// Requires: M % 128 == 0, N % 64 == 0, K % 32 == 0, lda/ldb % 4 == 0.
// ---------------------------------------------------------------------------
__global__ __launch_bounds__(256) void gemm_wmma_f16(
    const float* __restrict__ A, const float* __restrict__ B,
    float* __restrict__ C, int K, int lda, int ldb, int ldc, int Z1,
    long a1, long a2, long b1, long b2, long c1, long c2,
    int transB, float alpha, const float* __restrict__ bias, int relu)
{
  __shared__ alignas(16) _Float16 As[BM][BK + LPAD];
  __shared__ alignas(16) _Float16 Bs[BN][BK + LPAD];

  const int z  = blockIdx.z;
  const int z1 = z % Z1, z2 = z / Z1;
  A += (long)z1 * a1 + (long)z2 * a2;
  B += (long)z1 * b1 + (long)z2 * b2;
  C += (long)z1 * c1 + (long)z2 * c2;

  const int tid  = threadIdx.x;
  const int lane = tid & 31;
  const int wave = tid >> 5;
  const int hl = lane >> 4, ln = lane & 15;  // half-wave, lane-in-16
  const int wm = wave & 3,  wn = wave >> 2;  // 4x2 wave grid over 128x64 tile
  const int row0 = blockIdx.x * BM;
  const int col0 = blockIdx.y * BN;

  v8f acc[2][2] = {};

  for (int k0 = 0; k0 < K; k0 += BK) {
    __syncthreads();
    // ---- stage A tile 128x32: fp32 global -> f16 LDS (float4 loads) ----
    {
      const int kk = (tid & 7) * 4;
      const int m0 = tid >> 3;
#pragma unroll
      for (int p = 0; p < 4; ++p) {
        const int m = m0 + p * 32;
        const float4 f = *(const float4*)(A + (long)(row0 + m) * lda + (k0 + kk));
        As[m][kk + 0] = (_Float16)f.x; As[m][kk + 1] = (_Float16)f.y;
        As[m][kk + 2] = (_Float16)f.z; As[m][kk + 3] = (_Float16)f.w;
      }
    }
    // ---- stage B tile as Bs[n][kk] = B(k0+kk, col0+n) ----
    if (!transB) {
      const int n4  = (tid & 15) * 4;
      const int kk0 = tid >> 4;
#pragma unroll
      for (int p = 0; p < 2; ++p) {
        const int kk = kk0 + p * 16;
        const float4 f = *(const float4*)(B + (long)(k0 + kk) * ldb + (col0 + n4));
        Bs[n4 + 0][kk] = (_Float16)f.x; Bs[n4 + 1][kk] = (_Float16)f.y;
        Bs[n4 + 2][kk] = (_Float16)f.z; Bs[n4 + 3][kk] = (_Float16)f.w;
      }
    } else {  // B stored row-major N x K
      const int kk = (tid & 7) * 4;
      const int n0 = tid >> 3;
#pragma unroll
      for (int p = 0; p < 2; ++p) {
        const int n = n0 + p * 32;
        const float4 f = *(const float4*)(B + (long)(col0 + n) * ldb + (k0 + kk));
        Bs[n][kk + 0] = (_Float16)f.x; Bs[n][kk + 1] = (_Float16)f.y;
        Bs[n][kk + 2] = (_Float16)f.z; Bs[n][kk + 3] = (_Float16)f.w;
      }
    }
    __syncthreads();

    // ---- build fragments per ISA 7.12.2 layouts ----
    v16h af[2], bf[2];
#pragma unroll
    for (int i = 0; i < 2; ++i) {
      const int m = wm * 32 + i * 16 + ln;       // A: row = lane%16
      const h8 lo = *(const h8*)(&As[m][hl * 8]);        // K = hl*8 + 0..7
      const h8 hi = *(const h8*)(&As[m][16 + hl * 8]);   // K = 16 + hl*8 + 0..7
#pragma unroll
      for (int j = 0; j < 8; ++j) { af[i][j] = lo[j]; af[i][8 + j] = hi[j]; }
    }
#pragma unroll
    for (int j = 0; j < 2; ++j) {
      const int n = wn * 32 + j * 16 + ln;       // B: col = lane%16
      const h8 lo = *(const h8*)(&Bs[n][hl * 8]);        // K = hl*8 + 0..7
      const h8 hi = *(const h8*)(&Bs[n][16 + hl * 8]);   // K = 16 + hl*8 + 0..7
#pragma unroll
      for (int q = 0; q < 8; ++q) { bf[j][q] = lo[q]; bf[j][8 + q] = hi[q]; }
    }
#pragma unroll
    for (int i = 0; i < 2; ++i)
#pragma unroll
      for (int j = 0; j < 2; ++j)
        acc[i][j] = __builtin_amdgcn_wmma_f32_16x16x32_f16(
            false, af[i], false, bf[j], (short)0, acc[i][j], false, false);
  }

  // ---- epilogue: C/D layout -> row = 8*hl + r, col = lane%16 ----
#pragma unroll
  for (int i = 0; i < 2; ++i) {
    const int row = row0 + wm * 32 + i * 16 + hl * 8;
#pragma unroll
    for (int j = 0; j < 2; ++j) {
      const int col = col0 + wn * 32 + j * 16 + ln;
      const float bv = bias ? bias[col] : 0.0f;
#pragma unroll
      for (int r = 0; r < 8; ++r) {
        float v = acc[i][j][r] * alpha + bv;
        if (relu) v = fmaxf(v, 0.0f);
        C[(long)(row + r) * ldc + col] = v;
      }
    }
  }
}

// ---------------------------------------------------------------------------
// softmax over the DMODEL axis of (H, 1024, 64) weights with causal-tril mask:
// column (h,k): rows d < k are -inf -> output 0; rows d >= k softmaxed.
// One block per (h,k) column.
// ---------------------------------------------------------------------------
__global__ __launch_bounds__(128) void wmask_softmax(
    const float* __restrict__ W, float* __restrict__ O)
{
  const int DK = 64, DM = 1024;
  const int h = blockIdx.x >> 6;
  const int kc = blockIdx.x & 63;
  const float* base = W + (long)h * DM * DK + kc;
  float* ob = O + (long)h * DM * DK + kc;
  const int tid = threadIdx.x;
  __shared__ float red[128];

  float mx = -3.4e38f;
  for (int d = kc + tid; d < DM; d += 128) mx = fmaxf(mx, base[(long)d * DK]);
  red[tid] = mx; __syncthreads();
  for (int s = 64; s > 0; s >>= 1) {
    if (tid < s) red[tid] = fmaxf(red[tid], red[tid + s]);
    __syncthreads();
  }
  mx = red[0]; __syncthreads();

  float sum = 0.0f;
  for (int d = kc + tid; d < DM; d += 128) sum += expf(base[(long)d * DK] - mx);
  red[tid] = sum; __syncthreads();
  for (int s = 64; s > 0; s >>= 1) {
    if (tid < s) red[tid] += red[tid + s];
    __syncthreads();
  }
  const float inv = 1.0f / red[0];

  for (int d = tid; d < DM; d += 128)
    ob[(long)d * DK] = (d < kc) ? 0.0f : expf(base[(long)d * DK] - mx) * inv;
}

// ---------------------------------------------------------------------------
// Softmax over the QUERY axis (-2) of scores (Z, T, T): column softmax.
// Thread owns one column s; row loop is coalesced across the wave.
// ---------------------------------------------------------------------------
__global__ __launch_bounds__(256) void col_softmax(float* __restrict__ S)
{
  const int T = 1024;
  float* base = S + (long)blockIdx.y * T * T + blockIdx.x * 256 + threadIdx.x;
  float mx = -3.4e38f;
  for (int t = 0; t < T; ++t) mx = fmaxf(mx, base[(long)t * T]);
  float sum = 0.0f;
  for (int t = 0; t < T; ++t) sum += expf(base[(long)t * T] - mx);
  const float inv = 1.0f / sum;
  for (int t = 0; t < T; ++t) base[(long)t * T] = expf(base[(long)t * T] - mx) * inv;
}

// ---------------------------------------------------------------------------
// out = (v - mean(v)) / std(v, ddof=1), v = a + r, rows of length 1024
// ---------------------------------------------------------------------------
__global__ __launch_bounds__(256) void add_norm(
    const float* __restrict__ A, const float* __restrict__ R, float* __restrict__ O)
{
  const int D = 1024;
  const long base = (long)blockIdx.x * D;
  __shared__ float v[1024];
  __shared__ float r1[256], r2[256];
  const int tid = threadIdx.x;
  float s = 0.0f, s2 = 0.0f;
  for (int i = tid; i < D; i += 256) {
    const float x = A[base + i] + R[base + i];
    v[i] = x; s += x; s2 += x * x;
  }
  r1[tid] = s; r2[tid] = s2; __syncthreads();
  for (int st = 128; st > 0; st >>= 1) {
    if (tid < st) { r1[tid] += r1[tid + st]; r2[tid] += r2[tid + st]; }
    __syncthreads();
  }
  const float mean = r1[0] * (1.0f / D);
  const float var  = (r2[0] - (float)D * mean * mean) * (1.0f / (D - 1));
  const float istd = rsqrtf(var);
  for (int i = tid; i < D; i += 256) O[base + i] = (v[i] - mean) * istd;
}

// ---------------------------------------------------------------------------
// Host side
// ---------------------------------------------------------------------------
static inline void gemm(const float* A, const float* B, float* C,
                        int M, int N, int K, int lda, int ldb, int ldc,
                        int Z1, int Z2,
                        long a1, long a2, long b1, long b2, long c1, long c2,
                        int transB, float alpha, const float* bias, int relu,
                        hipStream_t s)
{
  dim3 grid(M / BM, N / BN, Z1 * Z2), blk(256);
  gemm_wmma_f16<<<grid, blk, 0, s>>>(A, B, C, K, lda, ldb, ldc, Z1,
                                     a1, a2, b1, b2, c1, c2,
                                     transB, alpha, bias, relu);
}

static void attention_block(const float* Qsrc, const float* KVsrc,
                            const float* Wq, const float* Wk, const float* Wv,
                            const float* Wo, const float* residual, float* outbuf,
                            float* Qb, float* Kb, float* Vb, float* Sc,
                            float* Cc, float* Ao, hipStream_t s)
{
  const int Bc = 4, Tc = 1024, Dc = 1024, Hc = 16, Kc = 64;
  const long hW = (long)Dc * Kc;        // per-head weight stride
  const long qS = (long)Tc * Kc;        // per-(b,h) Q/K/V stride
  const long sS = (long)Tc * Tc;        // per-(b,h) scores stride
  // Q/K/V projections, batched z = (h, b): M=T, N=64, K=1024
  gemm(Qsrc,  Wq, Qb, Tc, Kc, Dc, Dc, Kc, Kc, Hc, Bc,
       0, (long)Tc * Dc, hW, 0, qS, (long)Hc * qS, 0, 1.0f, nullptr, 0, s);
  gemm(KVsrc, Wk, Kb, Tc, Kc, Dc, Dc, Kc, Kc, Hc, Bc,
       0, (long)Tc * Dc, hW, 0, qS, (long)Hc * qS, 0, 1.0f, nullptr, 0, s);
  gemm(KVsrc, Wv, Vb, Tc, Kc, Dc, Dc, Kc, Kc, Hc, Bc,
       0, (long)Tc * Dc, hW, 0, qS, (long)Hc * qS, 0, 1.0f, nullptr, 0, s);
  // scores = Q K^T / sqrt(64): M=T, N=T, K=64, B transposed
  gemm(Qb, Kb, Sc, Tc, Tc, Kc, Kc, Kc, Tc, Hc, Bc,
       qS, (long)Hc * qS, qS, (long)Hc * qS, sS, (long)Hc * sS,
       1, 0.125f, nullptr, 0, s);
  // softmax over query axis (columns)
  col_softmax<<<dim3(Tc / 256, Bc * Hc), 256, 0, s>>>(Sc);
  // attn @ V -> written straight into head-concat layout (B, T, H*64)
  gemm(Sc, Vb, Cc, Tc, Kc, Tc, Tc, Kc, Dc, Hc, Bc,
       sS, (long)Hc * sS, qS, (long)Hc * qS, (long)Kc, (long)Tc * Dc,
       0, 1.0f, nullptr, 0, s);
  // output projection: (B*T, D) @ (D, D)
  gemm(Cc, Wo, Ao, Bc * Tc, Dc, Dc, Dc, Dc, Dc, 1, 1,
       0, 0, 0, 0, 0, 0, 0, 1.0f, nullptr, 0, s);
  add_norm<<<Bc * Tc, 256, 0, s>>>(Ao, residual, outbuf);
}

extern "C" void kernel_launch(void* const* d_in, const int* in_sizes, int n_in,
                              void* d_out, int out_size, void* d_ws, size_t ws_size,
                              hipStream_t stream)
{
  (void)in_sizes; (void)n_in; (void)out_size; (void)ws_size;
  const int Bc = 4, Tc = 1024, Dc = 1024, Hc = 16, Kc = 64, Fc = 4096;

  const float* x    = (const float*)d_in[0];
  const float* y    = (const float*)d_in[1];
  const float* Wq1  = (const float*)d_in[2];
  const float* Wk1  = (const float*)d_in[3];
  const float* Wv1  = (const float*)d_in[4];
  const float* Wo1  = (const float*)d_in[5];
  const float* Wq2  = (const float*)d_in[6];
  const float* Wk2  = (const float*)d_in[7];
  const float* Wv2  = (const float*)d_in[8];
  const float* Wo2  = (const float*)d_in[9];
  const float* w_in = (const float*)d_in[10];
  const float* b_in = (const float*)d_in[11];
  const float* w_out= (const float*)d_in[12];
  const float* b_out= (const float*)d_in[13];
  float* out = (float*)d_out;

  // workspace carve-up (floats)
  float* ws = (float*)d_ws;
  size_t o = 0;
  float* Wq1m = ws + o; o += (size_t)Hc * Dc * Kc;          // 1.05M
  float* Wk1m = ws + o; o += (size_t)Hc * Dc * Kc;
  float* Qb   = ws + o; o += (size_t)Bc * Hc * Tc * Kc;     // 4.19M each
  float* Kb   = ws + o; o += (size_t)Bc * Hc * Tc * Kc;
  float* Vb   = ws + o; o += (size_t)Bc * Hc * Tc * Kc;
  float* Sc   = ws + o; o += (size_t)Bc * Hc * Tc * Tc;     // 67.1M
  float* Cc   = ws + o; o += (size_t)Bc * Tc * Dc;
  float* Ao   = ws + o; o += (size_t)Bc * Tc * Dc;
  float* O1   = ws + o; o += (size_t)Bc * Tc * Dc;
  float* O2   = ws + o; o += (size_t)Bc * Tc * Dc;
  float* F1   = ws + o; o += (size_t)Bc * Tc * Fc;          // 16.8M

  // output tuple element 0: x passthrough
  hipMemcpyAsync(out, x, (size_t)Bc * Tc * Dc * sizeof(float),
                 hipMemcpyDeviceToDevice, stream);

  // layer 1: masked weight softmax then self-attention on y
  wmask_softmax<<<Hc * Kc, 128, 0, stream>>>(Wq1, Wq1m);
  wmask_softmax<<<Hc * Kc, 128, 0, stream>>>(Wk1, Wk1m);
  attention_block(y, y, Wq1m, Wk1m, Wv1, Wo1, y, O1,
                  Qb, Kb, Vb, Sc, Cc, Ao, stream);

  // layer 2: cross-attention (Q from out1, K/V from x), no weight mask
  attention_block(O1, x, Wq2, Wk2, Wv2, Wo2, O1, O2,
                  Qb, Kb, Vb, Sc, Cc, Ao, stream);

  // FFN: relu(out2 @ w_in^T + b_in) @ w_out^T + b_out, then add+norm
  gemm(O2, w_in, F1, Bc * Tc, Fc, Dc, Dc, Dc, Fc, 1, 1,
       0, 0, 0, 0, 0, 0, 1, 1.0f, b_in, 1, stream);
  gemm(F1, w_out, Ao, Bc * Tc, Dc, Fc, Fc, Fc, Dc, 1, 1,
       0, 0, 0, 0, 0, 0, 1, 1.0f, b_out, 0, stream);
  add_norm<<<Bc * Tc, 256, 0, stream>>>(Ao, O2, out + (size_t)Bc * Tc * Dc);
}
